// SlidingWindowAttention_42923903156553
// MI455X (gfx1250) — matrix operand order repaired and checked by reading, not compiled, not run on hardware
//
#include <hip/hip_runtime.h>

// ---------------------------------------------------------------------------
// Sliding-window attention for MI455X (gfx1250, wave32, WMMA bf16,
// async global->LDS copies double-buffered against the WMMA pipeline).
// B=2 S=4096 D=1024 H=16 Dh=64 W=256 nb=16
// ---------------------------------------------------------------------------

#define S_LEN 4096
#define D_DIM 1024
#define HEADS 16
#define DH 64
#define WIN 256
#define NB 16
#define BATCH 2

typedef __bf16 bf16_t;
typedef __attribute__((ext_vector_type(16))) __bf16 v16bf;
typedef __attribute__((ext_vector_type(8)))  __bf16 v8bf;
typedef __attribute__((ext_vector_type(4)))  __bf16 v4bf;
typedef __attribute__((ext_vector_type(8)))  float  v8f;

union FragU {
  v16bf v;
  struct { v8bf lo; v8bf hi; } p;
};

static __device__ __forceinline__ v16bf make_frag(v8bf lo, v8bf hi) {
  FragU u;
  u.p.lo = lo;
  u.p.hi = hi;
  return u.v;
}

static __device__ __forceinline__ v8f vsplat8(float x) {
  v8f v = {x, x, x, x, x, x, x, x};
  return v;
}

static __device__ __forceinline__ v8f wmma_bf16(v16bf a, v16bf b, v8f c) {
  // D = A(16x32 bf16) * B(32x16 bf16) + C(16x16 f32)
  return __builtin_amdgcn_wmma_f32_16x16x32_bf16(false, a, false, b,
                                                 (short)0, c, false, false);
}

static __device__ __forceinline__ bf16_t f2bf(float f) { return (bf16_t)f; }

// Async global->LDS 16-byte copy (ASYNCcnt-tracked, CDNA5 VGLOBAL op 98).
// lds_off: 32-bit LDS byte offset (low 32 bits of the generic LDS address).
static __device__ __forceinline__ void async_b128(unsigned lds_off,
                                                  const void* gptr) {
  const unsigned long long ga = (unsigned long long)(size_t)gptr;
  asm volatile("global_load_async_to_lds_b128 %0, %1, off"
               :
               : "v"(lds_off), "v"(ga)
               : "memory");
}

static __device__ __forceinline__ unsigned lds_addr(const void* p) {
  return (unsigned)(size_t)p;  // generic LDS address: low 32 bits = LDS offset
}

// ---------------------------------------------------------------------------
// Kernel 0: elementwise f32 -> bf16 (x and the four weight matrices).
// ---------------------------------------------------------------------------
__global__ __launch_bounds__(256) void swa_cvt_bf16(
    const float* __restrict__ in, bf16_t* __restrict__ out) {
  const size_t i = ((size_t)blockIdx.x * 256 + threadIdx.x) * 4;
  const float4 f = *(const float4*)(in + i);
  v4bf o = {f2bf(f.x), f2bf(f.y), f2bf(f.z), f2bf(f.w)};
  *(v4bf*)(out + i) = o;
}

// ---------------------------------------------------------------------------
// Unified bf16 GEMM:  acc = A[M,1024] @ Bm[1024,1024]^T (Bm row-major [n][k]).
// Block tile 128x128, 8 waves x (32x64), K-step 32, double-buffered LDS fed
// by async global->LDS b128 copies (4 per thread per stage).
// MODE 0: store bf16 permuted to head-major [B,H,S,Dh] (QKV projections).
// MODE 1: store f32 row-major [M,1024] (output projection).
// ---------------------------------------------------------------------------
template <int MODE>
__global__ __launch_bounds__(256) void swa_gemm_bf16(
    const bf16_t* __restrict__ A, const bf16_t* __restrict__ Bm,
    void* __restrict__ OutP) {
  __shared__ bf16_t At[2][128 * 40];  // [m][k], stride 40 bf16 (80B, 16B mult)
  __shared__ bf16_t Bt[2][128 * 40];  // [n][k]

  const int tid = threadIdx.x;
  const int lane = tid & 31;
  const int wid = tid >> 5;
  const int half = lane >> 4;
  const int l15 = lane & 15;

  const int m_base = blockIdx.y * 128;
  const int n_base = blockIdx.x * 128;
  const int wm0 = (wid & 3) * 32;
  const int wn0 = (wid >> 2) * 64;

  v8f acc[2][4];
#pragma unroll
  for (int i = 0; i < 2; ++i)
#pragma unroll
    for (int jj = 0; jj < 4; ++jj) acc[i][jj] = vsplat8(0.0f);

  // Issue one stage of async copies: 128x32 bf16 for A and B tiles.
  auto issue = [&](int step, int buf) {
    const int k0 = step * 32;
#pragma unroll
    for (int it = 0; it < 2; ++it) {
      const int slot = tid + it * 256;  // 0..511 16B slots
      const int r = slot >> 2;
      const int c = (slot & 3) * 8;
      async_b128(lds_addr(&At[buf][r * 40 + c]),
                 A + (size_t)(m_base + r) * D_DIM + k0 + c);
      async_b128(lds_addr(&Bt[buf][r * 40 + c]),
                 Bm + (size_t)(n_base + r) * D_DIM + k0 + c);
    }
  };

  const int KSTEPS = D_DIM / 32;  // 32
  issue(0, 0);
  issue(1, 1);

  for (int ks = 0; ks < KSTEPS; ++ks) {
    // Wait for stage ks (stage ks+1's 4 copies may stay in flight).
    if (ks + 1 < KSTEPS)
      asm volatile("s_wait_asynccnt 0x4" ::: "memory");
    else
      asm volatile("s_wait_asynccnt 0x0" ::: "memory");
    __syncthreads();

    const bf16_t* Ab = At[ks & 1];
    const bf16_t* Bb = Bt[ks & 1];

    v16bf af[2];
#pragma unroll
    for (int tm = 0; tm < 2; ++tm) {
      const bf16_t* p = &Ab[(wm0 + tm * 16 + l15) * 40];
      af[tm] = make_frag(*(const v8bf*)(p + half * 8),
                         *(const v8bf*)(p + 16 + half * 8));
    }
#pragma unroll
    for (int tn = 0; tn < 4; ++tn) {
      const bf16_t* p = &Bb[(wn0 + tn * 16 + l15) * 40];
      v16bf bfrag = make_frag(*(const v8bf*)(p + half * 8),
                              *(const v8bf*)(p + 16 + half * 8));
#pragma unroll
      for (int tm = 0; tm < 2; ++tm)
        acc[tm][tn] = wmma_bf16(af[tm], bfrag, acc[tm][tn]);
    }

    __syncthreads();  // all waves done with buf (ks&1) before overwrite
    if (ks + 2 < KSTEPS) issue(ks + 2, ks & 1);
  }

#pragma unroll
  for (int tm = 0; tm < 2; ++tm)
#pragma unroll
    for (int tn = 0; tn < 4; ++tn)
#pragma unroll
      for (int i = 0; i < 8; ++i) {
        const int gm = m_base + wm0 + tm * 16 + i + half * 8;
        const int gn = n_base + wn0 + tn * 16 + l15;
        if (MODE == 0) {
          const int bb = gm >> 12, ss = gm & (S_LEN - 1);
          const int hh = gn >> 6, dd = gn & (DH - 1);
          ((bf16_t*)OutP)[(((size_t)(bb * HEADS + hh) * S_LEN + ss) << 6) +
                          dd] = f2bf(acc[tm][tn][i]);
        } else {
          ((float*)OutP)[(size_t)gm * D_DIM + gn] = acc[tm][tn][i];
        }
      }
}

// ---------------------------------------------------------------------------
// Kernel 2: flash-style sliding-window attention.
// One workgroup per (b,h,j) query block of 256 rows; 8 waves x 32 rows.
// Streams 8 key chunks of 64 over the 2W=512 window with online softmax.
// K tile arrives via async global->LDS; V is transposed through VGPRs.
// ---------------------------------------------------------------------------
__global__ __launch_bounds__(256) void swa_attn(
    const bf16_t* __restrict__ Q, const bf16_t* __restrict__ Kb,
    const bf16_t* __restrict__ Vb, bf16_t* __restrict__ O) {
  __shared__ bf16_t Kt[64 * 72];      // [key][dh]   (B operand for scores)
  __shared__ bf16_t Vt[64 * 72];      // [dh][key]   (B operand for P@V)
  __shared__ bf16_t Pw[8 * 32 * 40];  // per-wave P scratch [row][key]

  const int tid = threadIdx.x;
  const int lane = tid & 31;
  const int wid = tid >> 5;
  const int half = lane >> 4;
  const int l15 = lane & 15;

  const int j = blockIdx.x & 15;
  const int h = (blockIdx.x >> 4) & 15;
  const int b = (int)(blockIdx.x >> 8);
  const size_t head = ((size_t)(b * HEADS + h)) * S_LEN * DH;

  // Q fragments (A operand, 32 rows x Dh=64 -> [tm][kk]) kept in registers.
  v16bf qf[2][2];
#pragma unroll
  for (int tm = 0; tm < 2; ++tm)
#pragma unroll
    for (int kk = 0; kk < 2; ++kk) {
      const bf16_t* qp =
          Q + head + (size_t)(j * WIN + wid * 32 + tm * 16 + l15) * DH + kk * 32;
      qf[tm][kk] = make_frag(*(const v8bf*)(qp + half * 8),
                             *(const v8bf*)(qp + 16 + half * 8));
    }

  v8f o_acc[2][4];
  v8f m_st[2], l_st[2];
#pragma unroll
  for (int tm = 0; tm < 2; ++tm) {
    m_st[tm] = vsplat8(-1e30f);
    l_st[tm] = vsplat8(0.0f);
#pragma unroll
    for (int tn = 0; tn < 4; ++tn) o_acc[tm][tn] = vsplat8(0.0f);
  }

  bf16_t* pw = &Pw[wid * 32 * 40];

  for (int kc = 0; kc < 8; ++kc) {
    __syncthreads();
    {  // cooperative load of 64 keys: K async to LDS, V transposed via VGPRs
      const int r = tid >> 2;
      const int seg = (tid & 3) * 16;
      int kr = (j - 1) * WIN + kc * 64 + r;
      kr = kr < 0 ? 0 : kr;  // invalid rows are fully masked below
      const bf16_t* kp = Kb + head + (size_t)kr * DH + seg;
      async_b128(lds_addr(&Kt[r * 72 + seg]), kp);
      async_b128(lds_addr(&Kt[r * 72 + seg + 8]), kp + 8);
      const bf16_t* vp = Vb + head + (size_t)kr * DH + seg;
      v8bf v0 = *(const v8bf*)vp;
      v8bf v1 = *(const v8bf*)(vp + 8);
#pragma unroll
      for (int c = 0; c < 8; ++c) {
        Vt[(seg + c) * 72 + r] = v0[c];
        Vt[(seg + 8 + c) * 72 + r] = v1[c];
      }
      asm volatile("s_wait_asynccnt 0x0" ::: "memory");
    }
    __syncthreads();

    // S = Q K^T for this chunk: [32 x 64] per wave (K-dim = Dh = 64).
    v8f sc[2][4];
#pragma unroll
    for (int tm = 0; tm < 2; ++tm)
#pragma unroll
      for (int tn = 0; tn < 4; ++tn) sc[tm][tn] = vsplat8(0.0f);
#pragma unroll
    for (int tn = 0; tn < 4; ++tn)
#pragma unroll
      for (int kk = 0; kk < 2; ++kk) {
        const bf16_t* p = &Kt[(tn * 16 + l15) * 72 + kk * 32 + half * 8];
        v16bf kf = make_frag(*(const v8bf*)p, *(const v8bf*)(p + 16));
#pragma unroll
        for (int tm = 0; tm < 2; ++tm)
          sc[tm][tn] = wmma_bf16(qf[tm][kk], kf, sc[tm][tn]);
      }

    // Online softmax update (per tm half-tile; rows mapped per C-layout).
#pragma unroll
    for (int tm = 0; tm < 2; ++tm) {
#pragma unroll
      for (int tn = 0; tn < 4; ++tn)
#pragma unroll
        for (int i = 0; i < 8; ++i) {
          const int qi = wid * 32 + tm * 16 + i + half * 8;
          const int ki = kc * 64 + tn * 16 + l15;
          const bool ok =
              (ki >= qi + 1) && (ki <= qi + WIN) && ((j > 0) || (ki >= WIN));
          sc[tm][tn][i] = ok ? sc[tm][tn][i] * 0.125f : -1e30f;
        }
      v8f rmx = sc[tm][0];
#pragma unroll
      for (int tn = 1; tn < 4; ++tn)
#pragma unroll
        for (int i = 0; i < 8; ++i) rmx[i] = fmaxf(rmx[i], sc[tm][tn][i]);
#pragma unroll
      for (int off = 1; off < 16; off <<= 1)
#pragma unroll
        for (int i = 0; i < 8; ++i)
          rmx[i] = fmaxf(rmx[i], __shfl_xor(rmx[i], off, 32));

      v8f mnew, alpha;
#pragma unroll
      for (int i = 0; i < 8; ++i) {
        mnew[i] = fmaxf(m_st[tm][i], rmx[i]);
        alpha[i] =
            (m_st[tm][i] > -1e29f) ? __expf(m_st[tm][i] - mnew[i]) : 0.0f;
      }
      m_st[tm] = mnew;

      v8f rsum = vsplat8(0.0f);
#pragma unroll
      for (int tn = 0; tn < 4; ++tn)
#pragma unroll
        for (int i = 0; i < 8; ++i) {
          const float p =
              (sc[tm][tn][i] > -1e29f) ? __expf(sc[tm][tn][i] - mnew[i]) : 0.0f;
          rsum[i] += p;
          pw[(tm * 16 + i + half * 8) * 40 + tn * 16 + l15] = f2bf(p);
        }
#pragma unroll
      for (int off = 1; off < 16; off <<= 1)
#pragma unroll
        for (int i = 0; i < 8; ++i) rsum[i] += __shfl_xor(rsum[i], off, 32);
      l_st[tm] = l_st[tm] * alpha + rsum;
#pragma unroll
      for (int tn = 0; tn < 4; ++tn) o_acc[tm][tn] *= alpha;
    }

    // Per-wave LDS scratch: make this wave's P stores visible before reload.
    asm volatile("s_wait_dscnt 0x0" ::: "memory");

    v16bf pf[2][2];
#pragma unroll
    for (int tm = 0; tm < 2; ++tm)
#pragma unroll
      for (int kk = 0; kk < 2; ++kk) {
        const bf16_t* p = &pw[(tm * 16 + l15) * 40 + kk * 32];
        pf[tm][kk] = make_frag(*(const v8bf*)(p + half * 8),
                               *(const v8bf*)(p + 16 + half * 8));
      }
#pragma unroll
    for (int tn = 0; tn < 4; ++tn)
#pragma unroll
      for (int kk = 0; kk < 2; ++kk) {
        const bf16_t* p = &Vt[(tn * 16 + l15) * 72 + kk * 32 + half * 8];
        v16bf vf = make_frag(*(const v8bf*)p, *(const v8bf*)(p + 16));
#pragma unroll
        for (int tm = 0; tm < 2; ++tm)
          o_acc[tm][tn] = wmma_bf16(pf[tm][kk], vf, o_acc[tm][tn]);
      }
  }

  // Normalize and store bf16 attention output, layout [B*S, D] (col=h*64+dh).
#pragma unroll
  for (int tm = 0; tm < 2; ++tm)
#pragma unroll
    for (int tn = 0; tn < 4; ++tn)
#pragma unroll
      for (int i = 0; i < 8; ++i) {
        const int qrow = wid * 32 + tm * 16 + i + half * 8;
        const float ov = o_acc[tm][tn][i] / l_st[tm][i];
        O[((size_t)(b * S_LEN + j * WIN + qrow)) * D_DIM + h * DH + tn * 16 +
          l15] = f2bf(ov);
      }
}

// ---------------------------------------------------------------------------
extern "C" void kernel_launch(void* const* d_in, const int* in_sizes, int n_in,
                              void* d_out, int out_size, void* d_ws,
                              size_t ws_size, hipStream_t stream) {
  const float* x = (const float*)d_in[0];
  const float* wq = (const float*)d_in[1];
  const float* wk = (const float*)d_in[2];
  const float* wv = (const float*)d_in[3];
  const float* wo = (const float*)d_in[4];
  float* out = (float*)d_out;

  const size_t nx = (size_t)BATCH * S_LEN * D_DIM;  // 8,388,608
  const size_t nw = (size_t)D_DIM * D_DIM;          // 1,048,576
  const size_t per = (size_t)BATCH * HEADS * S_LEN * DH;  // 8,388,608

  // Workspace carve (~92 MB): bf16 copies of x/weights, head-major Q/K/V,
  // bf16 attention output.
  bf16_t* xb = (bf16_t*)d_ws;
  bf16_t* wqb = xb + nx;
  bf16_t* wkb = wqb + nw;
  bf16_t* wvb = wkb + nw;
  bf16_t* wob = wvb + nw;
  bf16_t* Qb = wob + nw;
  bf16_t* Kb = Qb + per;
  bf16_t* Vb = Kb + per;
  bf16_t* Ab = Vb + per;

  dim3 bb(256);
  swa_cvt_bf16<<<dim3((unsigned)(nx / 1024)), bb, 0, stream>>>(x, xb);
  swa_cvt_bf16<<<dim3((unsigned)(nw / 1024)), bb, 0, stream>>>(wq, wqb);
  swa_cvt_bf16<<<dim3((unsigned)(nw / 1024)), bb, 0, stream>>>(wk, wkb);
  swa_cvt_bf16<<<dim3((unsigned)(nw / 1024)), bb, 0, stream>>>(wv, wvb);
  swa_cvt_bf16<<<dim3((unsigned)(nw / 1024)), bb, 0, stream>>>(wo, wob);

  dim3 gg(D_DIM / 128, (BATCH * S_LEN) / 128);
  swa_gemm_bf16<0><<<gg, bb, 0, stream>>>(xb, wqb, (void*)Qb);
  swa_gemm_bf16<0><<<gg, bb, 0, stream>>>(xb, wkb, (void*)Kb);
  swa_gemm_bf16<0><<<gg, bb, 0, stream>>>(xb, wvb, (void*)Vb);
  swa_attn<<<dim3(BATCH * HEADS * NB), bb, 0, stream>>>(Qb, Kb, Vb, Ab);
  swa_gemm_bf16<1><<<gg, bb, 0, stream>>>(Ab, wob, (void*)out);
}
